// Model_61125974557083
// MI455X (gfx1250) — matrix operand order repaired and checked by reading, not compiled, not run on hardware
//
#include <hip/hip_runtime.h>
#include <hip/hip_bf16.h>
#include <math.h>

// ---------------------------------------------------------------------------
// Model dims
// ---------------------------------------------------------------------------
#define Bc    32
#define Lc    512
#define Nc    64
#define Dc    512
#define TDIMc 4
#define KERNc 8
#define Hnew  8
#define Dhc   64
#define LPc   63
#define PATCHc 16
#define STRIDEc 8
#define Sc    516        // D + TDIM tokens for inverted embedding
#define Henc  8
#define FFc   2048
#define PREDc 96

typedef __bf16 bf16_t;
typedef __attribute__((ext_vector_type(16))) __bf16 v16bf;
typedef __attribute__((ext_vector_type(8)))  __bf16 v8bf;
typedef __attribute__((ext_vector_type(4)))  __bf16 v4bf;
typedef __attribute__((ext_vector_type(8)))  float  v8f;

__device__ __forceinline__ float gelu_tanh(float x) {
  float c = 0.7978845608028654f * (x + 0.044715f * x * x * x);
  return 0.5f * x * (1.0f + tanhf(c));
}

// ---------------------------------------------------------------------------
// LDS-staged, double-buffered batched GEMM.
//   C[z] = act(alpha * A[z] @ B[z] + bias) (+ residual)
// Block: 256 thr / 8 waves -> 64(M) x 128(N) C macro-tile.
// Wave (wm,wn) owns 32x32 of C = 2x2 WMMA 16x16 tiles.
// Per K=32 step: stage A(64x32) + B(128x32 as [n][k]) to LDS bf16,
// barrier, each wave does 8x ds_load_b128 fragments + 4x v_wmma.
// Edge tiles use branchless clamped loads (address clamp + select-0),
// so no per-element EXEC branching / per-load waits.
// batch index z: zb = z / zdiv, zh = z % zdiv ; offsets zb*s? + zh*s?2
// transB: B element (k,n) read from Bm[n*ldb + k]  (i.e. x @ W.T)
// ---------------------------------------------------------------------------
__global__ __launch_bounds__(256) void gemm_bf16(
    const float* __restrict__ A, const float* __restrict__ Bm,
    float* __restrict__ C, const float* __restrict__ bias,
    const float* __restrict__ Res,
    int M, int N, int K, int lda, int ldb, int ldc,
    long sA, long sA2, long sB, long sB2, long sC, long sC2,
    int zdiv, int transB, float alpha, int act) {
  __shared__ bf16_t As[2][64][32];
  __shared__ bf16_t Bs[2][128][32];

  int tid  = threadIdx.x;
  int lane = tid & 31;
  int wave = tid >> 5;
  int hi   = lane >> 4;
  int l15  = lane & 15;

  int z  = blockIdx.z;
  int zb = z / zdiv;
  int zh = z - zb * zdiv;
  A  += (size_t)zb * sA + (size_t)zh * sA2;
  Bm += (size_t)zb * sB + (size_t)zh * sB2;
  C  += (size_t)zb * sC + (size_t)zh * sC2;
  if (Res) Res += (size_t)zb * sC + (size_t)zh * sC2;

  int m0 = blockIdx.y * 64;
  int n0 = blockIdx.x * 128;
  int mrow0 = (wave >> 2) * 32;   // wave's M offset within macro-tile
  int nrow0 = (wave & 3) * 32;    // wave's N offset within macro-tile

  bool fullN = (n0 + 128 <= N);

  v8f acc[2][2] = {};

  for (int k0 = 0; k0 < K; k0 += 32) {
    int db = (k0 >> 5) & 1;
    bool fullK = (k0 + 32 <= K);

    // ---- stage A panel: 64 x 32, 512 float4-chunks, 2 per thread ----
#pragma unroll
    for (int it = 0; it < 2; ++it) {
      int q   = tid + it * 256;
      int row = q >> 3;
      int kc  = (q & 7) * 4;
      int gm  = m0 + row;
      v4bf pk;
      if (gm < M && fullK) {
        const float* src = A + (size_t)gm * (size_t)lda + k0 + kc;
        pk[0] = (bf16_t)src[0]; pk[1] = (bf16_t)src[1];
        pk[2] = (bf16_t)src[2]; pk[3] = (bf16_t)src[3];
      } else {
        int gmc = gm < M ? gm : (M - 1);           // clamped, always legal
        const float* rowp = A + (size_t)gmc * (size_t)lda;
#pragma unroll
        for (int j = 0; j < 4; ++j) {
          int gk  = k0 + kc + j;
          int gkc = gk < K ? gk : (K - 1);
          float x = rowp[gkc];                     // branchless load
          pk[j] = (bf16_t)((gm < M && gk < K) ? x : 0.0f);
        }
      }
      *(v4bf*)(&As[db][row][kc]) = pk;
    }

    // ---- stage B panel into [n][k]: 128 x 32, 1024 chunks, 4 per thread ----
    if (transB) {
#pragma unroll
      for (int it = 0; it < 4; ++it) {
        int q  = tid + it * 256;
        int n  = q >> 3;
        int kc = (q & 7) * 4;
        int gn = n0 + n;
        v4bf pk;
        if (gn < N && fullK) {
          const float* src = Bm + (size_t)gn * (size_t)ldb + k0 + kc;
          pk[0] = (bf16_t)src[0]; pk[1] = (bf16_t)src[1];
          pk[2] = (bf16_t)src[2]; pk[3] = (bf16_t)src[3];
        } else {
          int gnc = gn < N ? gn : (N - 1);
          const float* rowp = Bm + (size_t)gnc * (size_t)ldb;
#pragma unroll
          for (int j = 0; j < 4; ++j) {
            int gk  = k0 + kc + j;
            int gkc = gk < K ? gk : (K - 1);
            float x = rowp[gkc];
            pk[j] = (bf16_t)((gn < N && gk < K) ? x : 0.0f);
          }
        }
        *(v4bf*)(&Bs[db][n][kc]) = pk;
      }
    } else {
#pragma unroll
      for (int it = 0; it < 4; ++it) {
        int q  = tid + it * 256;
        int kk = q >> 5;
        int nc = (q & 31) * 4;
        int gk = k0 + kk;
        if (gk < K && fullN) {
          const float* src = Bm + (size_t)gk * (size_t)ldb + n0 + nc;
          Bs[db][nc + 0][kk] = (bf16_t)src[0];
          Bs[db][nc + 1][kk] = (bf16_t)src[1];
          Bs[db][nc + 2][kk] = (bf16_t)src[2];
          Bs[db][nc + 3][kk] = (bf16_t)src[3];
        } else {
          int gkc = gk < K ? gk : (K - 1);
          const float* rowp = Bm + (size_t)gkc * (size_t)ldb;
#pragma unroll
          for (int j = 0; j < 4; ++j) {
            int gn  = n0 + nc + j;
            int gnc = gn < N ? gn : (N - 1);
            float x = rowp[gnc];
            Bs[db][nc + j][kk] = (bf16_t)((gk < K && gn < N) ? x : 0.0f);
          }
        }
      }
    }

    __syncthreads();

    // ---- fragments from LDS (2x ds_load_b128 each) ----
    v16bf afr[2], bfr[2];
#pragma unroll
    for (int i = 0; i < 2; ++i) {
      const bf16_t* Ap = &As[db][mrow0 + i * 16 + l15][hi * 8];
      v8bf lo = *(const v8bf*)Ap;          // k = hi*8 .. +7      -> elems 0..7
      v8bf hh = *(const v8bf*)(Ap + 16);   // k = hi*8+16 .. +7   -> elems 8..15
#pragma unroll
      for (int e = 0; e < 8; ++e) { afr[i][e] = lo[e]; afr[i][8 + e] = hh[e]; }
    }
#pragma unroll
    for (int j = 0; j < 2; ++j) {
      const bf16_t* Bp = &Bs[db][nrow0 + j * 16 + l15][hi * 8];
      v8bf lo = *(const v8bf*)Bp;          // k = hi*8 .. +7      -> elems 0..7
      v8bf hh = *(const v8bf*)(Bp + 16);   // k = hi*8+16 .. +7   -> elems 8..15
#pragma unroll
      for (int e = 0; e < 8; ++e) { bfr[j][e] = lo[e]; bfr[j][8 + e] = hh[e]; }
    }

#pragma unroll
    for (int i = 0; i < 2; ++i)
#pragma unroll
      for (int j = 0; j < 2; ++j)
        acc[i][j] = __builtin_amdgcn_wmma_f32_16x16x32_bf16(
            false, afr[i], false, bfr[j], (short)0, acc[i][j], false, false);
  }

  // ---- epilogue ----
#pragma unroll
  for (int j = 0; j < 2; ++j) {
    int nn = n0 + nrow0 + j * 16 + l15;
    if (nn < N) {
      float bv = bias ? bias[nn] : 0.0f;
#pragma unroll
      for (int i = 0; i < 2; ++i) {
#pragma unroll
        for (int r = 0; r < 8; ++r) {
          int mm = m0 + mrow0 + i * 16 + r + hi * 8;
          if (mm < M) {
            float v = acc[i][j][r] * alpha + bv;
            if (act == 1) v = gelu_tanh(v);
            if (Res) v += Res[(size_t)mm * (size_t)ldc + nn];
            C[(size_t)mm * (size_t)ldc + nn] = v;
          }
        }
      }
    }
  }
}

// ---------------------------------------------------------------------------
// Elementwise / reduction kernels
// ---------------------------------------------------------------------------
__global__ void stats_k(const float* __restrict__ x, float* __restrict__ mean,
                        float* __restrict__ stdv, int total) {
  int i = blockIdx.x * blockDim.x + threadIdx.x;
  if (i >= total) return;
  int b = i / Nc, n = i - b * Nc;
  const float* p = x + (size_t)b * Lc * Nc + n;
  float s = 0.0f;
  for (int l = 0; l < Lc; ++l) s += p[(size_t)l * Nc];
  float m = s / (float)Lc;
  float v = 0.0f;
  for (int l = 0; l < Lc; ++l) { float d = p[(size_t)l * Nc] - m; v += d * d; }
  v /= (float)Lc;
  mean[i] = m;
  stdv[i] = sqrtf(v + 1e-5f);
}

// im2col for circular Conv1d(N->D, k=3): Ac[b, l, k*Nc+n] = xn[b, (l+k-1)%L, n]
__global__ void aconv_k(const float* __restrict__ x, const float* __restrict__ mean,
                        const float* __restrict__ stdv, float* __restrict__ Ac, int total) {
  int i = blockIdx.x * blockDim.x + threadIdx.x;
  if (i >= total) return;
  int kk = i % (3 * Nc);
  int l  = (i / (3 * Nc)) % Lc;
  int b  = i / (3 * Nc * Lc);
  int k = kk / Nc, n = kk - k * Nc;
  int lc = l + k - 1;
  if (lc < 0) lc += Lc;
  if (lc >= Lc) lc -= Lc;
  float val = (x[(size_t)b * Lc * Nc + (size_t)lc * Nc + n] - mean[b * Nc + n]) / stdv[b * Nc + n];
  Ac[i] = val;
}

// Wt[k*Nc+n, d] = tok_W[d, n, k]   (tok_W: [D, N, 3])
__global__ void wtok_k(const float* __restrict__ w, float* __restrict__ Wt, int total) {
  int i = blockIdx.x * blockDim.x + threadIdx.x;
  if (i >= total) return;
  int d  = i % Dc;
  int kk = i / Dc;
  int k = kk / Nc, n = kk - k * Nc;
  Wt[i] = w[(size_t)d * Nc * 3 + (size_t)n * 3 + k];
}

// Wp[t*Dc+din, dout] = patch_W[dout, din, t]   (patch_W: [D, D, 16])
__global__ void wpat_k(const float* __restrict__ w, float* __restrict__ Wp, int total) {
  int i = blockIdx.x * blockDim.x + threadIdx.x;
  if (i >= total) return;
  int dout = i % Dc;
  int kk   = i / Dc;
  int t = kk / Dc, din = kk - t * Dc;
  Wp[i] = w[(size_t)dout * Dc * PATCHc + (size_t)din * PATCHc + t];
}

// tp[b, lp, t] = mean_j x_mark[b, lp*8 + j, t]
__global__ void tp_k(const float* __restrict__ xm, float* __restrict__ tp, int total) {
  int i = blockIdx.x * blockDim.x + threadIdx.x;
  if (i >= total) return;
  int t  = i % TDIMc;
  int lp = (i / TDIMc) % LPc;
  int b  = i / (TDIMc * LPc);
  float s = 0.0f;
  for (int j = 0; j < PATCHc; ++j)
    s += xm[(size_t)b * Lc * TDIMc + (size_t)(lp * STRIDEc + j) * TDIMc + t];
  tp[i] = s * (1.0f / (float)PATCHc);
}

// xg[b,h,l,dh] = (sum_{t,k} w*exp(-(tp-mu)^2/(2 sig^2))) * xp[b,l,h*64+dh] + gamma[h,dh]
__global__ void gauss_k(const float* __restrict__ xp, const float* __restrict__ tp,
                        const float* __restrict__ mu, const float* __restrict__ sg,
                        const float* __restrict__ w,  const float* __restrict__ gm,
                        float* __restrict__ xg, int total) {
  int i = blockIdx.x * blockDim.x + threadIdx.x;
  if (i >= total) return;
  int dh = i % Dhc;
  int l  = (i / Dhc) % LPc;
  int h  = (i / (Dhc * LPc)) % Hnew;
  int b  = i / (Dhc * LPc * Hnew);
  float tt[TDIMc];
#pragma unroll
  for (int t = 0; t < TDIMc; ++t) tt[t] = tp[((size_t)(b * LPc + l)) * TDIMc + t];
  size_t poff = (size_t)(h * Dhc + dh) * (TDIMc * KERNc);
  const float* muP = mu + poff;
  const float* sgP = sg + poff;
  const float* wP  = w + poff;
  float acc = 0.0f;
#pragma unroll
  for (int t = 0; t < TDIMc; ++t) {
#pragma unroll
    for (int k = 0; k < KERNc; ++k) {
      int j = t * KERNc + k;
      float d = tt[t] - muP[j];
      float s = sgP[j];
      acc += wP[j] * __expf(-(d * d) / (2.0f * s * s));
    }
  }
  float ph = xp[((size_t)(b * LPc + l)) * Dc + h * Dhc + dh];
  xg[i] = acc * ph + gm[h * Dhc + dh];
}

// In-place row softmax; one block per row
__global__ void softmax_k(float* __restrict__ X, int ld, int len, int rows) {
  int row = blockIdx.x;
  if (row >= rows) return;
  float* p = X + (size_t)row * (size_t)ld;
  __shared__ float red[256];
  int tid = threadIdx.x, nt = blockDim.x;
  float m = -3.0e38f;
  for (int i = tid; i < len; i += nt) m = fmaxf(m, p[i]);
  red[tid] = m; __syncthreads();
  for (int s = nt >> 1; s > 0; s >>= 1) {
    if (tid < s) red[tid] = fmaxf(red[tid], red[tid + s]);
    __syncthreads();
  }
  m = red[0]; __syncthreads();
  float sum = 0.0f;
  for (int i = tid; i < len; i += nt) { float e = __expf(p[i] - m); p[i] = e; sum += e; }
  red[tid] = sum; __syncthreads();
  for (int s = nt >> 1; s > 0; s >>= 1) {
    if (tid < s) red[tid] += red[tid + s];
    __syncthreads();
  }
  float inv = 1.0f / red[0];
  for (int i = tid; i < len; i += nt) p[i] *= inv;
}

// LayerNorm over last dim; one block per row
__global__ void layernorm_k(const float* __restrict__ X, const float* __restrict__ g,
                            const float* __restrict__ b, float* __restrict__ Y,
                            int D, int rows) {
  int row = blockIdx.x;
  if (row >= rows) return;
  const float* x = X + (size_t)row * (size_t)D;
  float* y = Y + (size_t)row * (size_t)D;
  __shared__ float red[256];
  int tid = threadIdx.x, nt = blockDim.x;
  float s = 0.0f;
  for (int i = tid; i < D; i += nt) s += x[i];
  red[tid] = s; __syncthreads();
  for (int st = nt >> 1; st > 0; st >>= 1) { if (tid < st) red[tid] += red[tid + st]; __syncthreads(); }
  float mean = red[0] / (float)D; __syncthreads();
  float v = 0.0f;
  for (int i = tid; i < D; i += nt) { float d = x[i] - mean; v += d * d; }
  red[tid] = v; __syncthreads();
  for (int st = nt >> 1; st > 0; st >>= 1) { if (tid < st) red[tid] += red[tid + st]; __syncthreads(); }
  float rs = rsqrtf(red[0] / (float)D + 1e-5f);
  for (int i = tid; i < D; i += nt) y[i] = (x[i] - mean) * rs * g[i] + b[i];
}

// zpre[b, c, lp] : c<512 -> xp[b,lp,c] ; else tp[b,lp,c-512]
__global__ void zpre_k(const float* __restrict__ xp, const float* __restrict__ tp,
                       float* __restrict__ zp, int total) {
  int i = blockIdx.x * blockDim.x + threadIdx.x;
  if (i >= total) return;
  int lp = i % LPc;
  int c  = (i / LPc) % Sc;
  int b  = i / (LPc * Sc);
  float v;
  if (c < Dc) v = xp[((size_t)(b * LPc + lp)) * Dc + c];
  else        v = tp[((size_t)(b * LPc + lp)) * TDIMc + (c - Dc)];
  zp[i] = v;
}

// out[b, pr, n] = po[b, n, pr] * stdv[b,n] + mean[b,n]  (po: [B, 516, 96])
__global__ void final_k(const float* __restrict__ po, const float* __restrict__ mean,
                        const float* __restrict__ stdv, float* __restrict__ out, int total) {
  int i = blockIdx.x * blockDim.x + threadIdx.x;
  if (i >= total) return;
  int n  = i % Nc;
  int pr = (i / Nc) % PREDc;
  int b  = i / (Nc * PREDc);
  float v = po[(size_t)b * Sc * PREDc + (size_t)n * PREDc + pr];
  out[i] = v * stdv[b * Nc + n] + mean[b * Nc + n];
}

// ---------------------------------------------------------------------------
// Host side
// ---------------------------------------------------------------------------
static inline int cdiv(int a, int b) { return (a + b - 1) / b; }

static void launch_gemm(hipStream_t st, const float* A, const float* B, float* C,
                        const float* bias, const float* res,
                        int M, int N, int K, int lda, int ldb, int ldc,
                        long sA, long sA2, long sB, long sB2, long sC, long sC2,
                        int batch, int zdiv, int transB, float alpha, int act) {
  dim3 grid(cdiv(N, 128), cdiv(M, 64), batch);
  gemm_bf16<<<grid, dim3(256), 0, st>>>(A, B, C, bias, res, M, N, K, lda, ldb, ldc,
                                        sA, sA2, sB, sB2, sC, sC2, zdiv, transB, alpha, act);
}

extern "C" void kernel_launch(void* const* d_in, const int* in_sizes, int n_in,
                              void* d_out, int out_size, void* d_ws, size_t ws_size,
                              hipStream_t stream) {
  (void)in_sizes; (void)n_in; (void)out_size; (void)ws_size;
  const float* x_enc  = (const float*)d_in[0];
  const float* x_mark = (const float*)d_in[1];
  // d_in[2], d_in[3] (x_dec, x_mark_dec) unused by forecast
  const float* tok_W   = (const float*)d_in[4];
  const float* patch_W = (const float*)d_in[5];
  const float* patch_b = (const float*)d_in[6];
  // new_layers: base 7, 17 tensors each (mu/sigma/w/gamma/W per Q,K,V then W_O,b_O)
  const int NEW0 = 7;
  const float* inv_W = (const float*)d_in[41];
  const float* inv_b = (const float*)d_in[42];
  const int ENC0 = 43;  // 16 tensors each: Wq,bq,Wk,bk,Wv,bv,Wo,bo,W1,b1,W2,b2,ln1g,ln1b,ln2g,ln2b
  const float* norm_g = (const float*)d_in[75];
  const float* norm_b = (const float*)d_in[76];
  const float* proj_W = (const float*)d_in[77];
  const float* proj_b = (const float*)d_in[78];

  float* ws = (float*)d_ws;
  size_t o = 0;
  float* MEAN  = ws + o; o += (size_t)Bc * Nc;
  float* STDV  = ws + o; o += (size_t)Bc * Nc;
  float* WTOK  = ws + o; o += (size_t)3 * Nc * Dc;              // [192, 512]
  float* ACONV = ws + o; o += (size_t)Bc * Lc * 3 * Nc;         // [B, 512, 192]
  float* Hbuf  = ws + o; o += (size_t)Bc * Lc * Dc;             // [B, 512, 512]
  float* WPAT  = ws + o; o += (size_t)PATCHc * Dc * Dc;         // [8192, 512]
  float* XP0   = ws + o; o += (size_t)Bc * LPc * Dc;
  float* XP1   = ws + o; o += (size_t)Bc * LPc * Dc;
  float* TP    = ws + o; o += (size_t)Bc * LPc * TDIMc;
  float* XG    = ws + o; o += (size_t)Bc * Hnew * LPc * Dhc;
  float* QQ    = ws + o; o += (size_t)Bc * Hnew * LPc * Dhc;
  float* KK    = ws + o; o += (size_t)Bc * Hnew * LPc * Dhc;
  float* VV    = ws + o; o += (size_t)Bc * Hnew * LPc * Dhc;
  float* TSC   = ws + o; o += (size_t)Bc * Hnew * LPc * LPc;
  float* OUT2  = ws + o; o += (size_t)Bc * LPc * Dc;
  float* ZPRE  = ws + o; o += (size_t)Bc * Sc * LPc;
  float* Z     = ws + o; o += (size_t)Bc * Sc * Dc;
  float* Z2    = ws + o; o += (size_t)Bc * Sc * Dc;
  float* QB    = ws + o; o += (size_t)Sc * Dc;
  float* KB    = ws + o; o += (size_t)Sc * Dc;
  float* VB    = ws + o; o += (size_t)Sc * Dc;
  float* CTXB  = ws + o; o += (size_t)Sc * Dc;
  float* ESC   = ws + o; o += (size_t)Henc * Sc * Sc;
  float* FFB   = ws + o; o += (size_t)Sc * FFc;
  float* PO    = ws + o; o += (size_t)Bc * Sc * PREDc;

  // ---- stats + token embedding (circular conv as GEMM) ----
  {
    int tot = Bc * Nc;
    stats_k<<<cdiv(tot, 256), 256, 0, stream>>>(x_enc, MEAN, STDV, tot);
  }
  {
    int tot = 3 * Nc * Dc;
    wtok_k<<<cdiv(tot, 256), 256, 0, stream>>>(tok_W, WTOK, tot);
  }
  {
    int tot = PATCHc * Dc * Dc;
    wpat_k<<<cdiv(tot, 256), 256, 0, stream>>>(patch_W, WPAT, tot);
  }
  {
    int tot = Bc * Lc * 3 * Nc;
    aconv_k<<<cdiv(tot, 256), 256, 0, stream>>>(x_enc, MEAN, STDV, ACONV, tot);
  }
  {
    int tot = Bc * LPc * TDIMc;
    tp_k<<<cdiv(tot, 256), 256, 0, stream>>>(x_mark, TP, tot);
  }
  // h = Aconv @ Wtok : [B] x [512,192] @ [192,512]
  launch_gemm(stream, ACONV, WTOK, Hbuf, nullptr, nullptr,
              Lc, Dc, 3 * Nc, 3 * Nc, Dc, Dc,
              (long)Lc * 3 * Nc, 0, 0, 0, (long)Lc * Dc, 0,
              Bc, 1, 0, 1.0f, 0);
  // patch conv as GEMM on overlapping rows of h: row lp starts at h + lp*8*512, K=8192
  launch_gemm(stream, Hbuf, WPAT, XP0, patch_b, nullptr,
              LPc, Dc, PATCHc * Dc, STRIDEc * Dc, Dc, Dc,
              (long)Lc * Dc, 0, 0, 0, (long)LPc * Dc, 0,
              Bc, 1, 0, 1.0f, 0);

  // ---- TimeKernelAttention layers ----
  float* xin = XP0;
  float* xout = XP1;
  float* qkv[3] = {QQ, KK, VV};
  for (int li = 0; li < 2; ++li) {
    int base = NEW0 + li * 17;
    for (int nm = 0; nm < 3; ++nm) {
      const float* mu = (const float*)d_in[base + nm * 5 + 0];
      const float* sg = (const float*)d_in[base + nm * 5 + 1];
      const float* w  = (const float*)d_in[base + nm * 5 + 2];
      const float* gm = (const float*)d_in[base + nm * 5 + 3];
      const float* Wp = (const float*)d_in[base + nm * 5 + 4];   // [H, Dh, Dh]
      int tot = Bc * Hnew * LPc * Dhc;
      gauss_k<<<cdiv(tot, 256), 256, 0, stream>>>(xin, TP, mu, sg, w, gm, XG, tot);
      // Q/K/V = xg @ W[h] : batch (b,h), M=63, K=64, N=64
      launch_gemm(stream, XG, Wp, qkv[nm], nullptr, nullptr,
                  LPc, Dhc, Dhc, Dhc, Dhc, Dhc,
                  (long)Hnew * LPc * Dhc, (long)LPc * Dhc,
                  0, (long)Dhc * Dhc,
                  (long)Hnew * LPc * Dhc, (long)LPc * Dhc,
                  Bc * Hnew, Hnew, 0, 1.0f, 0);
    }
    // scores = Q @ K^T / 8
    launch_gemm(stream, QQ, KK, TSC, nullptr, nullptr,
                LPc, LPc, Dhc, Dhc, Dhc, LPc,
                (long)Hnew * LPc * Dhc, (long)LPc * Dhc,
                (long)Hnew * LPc * Dhc, (long)LPc * Dhc,
                (long)Hnew * LPc * LPc, (long)LPc * LPc,
                Bc * Hnew, Hnew, 1, 0.125f, 0);
    softmax_k<<<Bc * Hnew * LPc, 64, 0, stream>>>(TSC, LPc, LPc, Bc * Hnew * LPc);
    // ctx = A @ V, written head-interleaved into [B, LP, 512]
    launch_gemm(stream, TSC, VV, OUT2, nullptr, nullptr,
                LPc, Dhc, LPc, LPc, Dhc, Dc,
                (long)Hnew * LPc * LPc, (long)LPc * LPc,
                (long)Hnew * LPc * Dhc, (long)LPc * Dhc,
                (long)LPc * Dc, (long)Dhc,
                Bc * Hnew, Hnew, 0, 1.0f, 0);
    // out = ctx @ W_O^T + b_O
    const float* WO = (const float*)d_in[base + 15];
    const float* bO = (const float*)d_in[base + 16];
    launch_gemm(stream, OUT2, WO, xout, bO, nullptr,
                LPc, Dc, Dc, Dc, Dc, Dc,
                (long)LPc * Dc, 0, 0, 0, (long)LPc * Dc, 0,
                Bc, 1, 1, 1.0f, 0);
    float* t = xin; xin = xout; xout = t;
  }

  // ---- inverted embedding: z = concat(xp^T, tp^T) @ inv_W^T + inv_b ----
  {
    int tot = Bc * Sc * LPc;
    zpre_k<<<cdiv(tot, 256), 256, 0, stream>>>(xin, TP, ZPRE, tot);
  }
  launch_gemm(stream, ZPRE, inv_W, Z, inv_b, nullptr,
              Sc, Dc, LPc, LPc, LPc, Dc,
              (long)Sc * LPc, 0, 0, 0, (long)Sc * Dc, 0,
              Bc, 1, 1, 1.0f, 0);

  // ---- encoder layers (full attention over S=516, 8 heads) ----
  for (int el = 0; el < 2; ++el) {
    int e0 = ENC0 + el * 16;
    const float* Wq = (const float*)d_in[e0 + 0];
    const float* bq = (const float*)d_in[e0 + 1];
    const float* Wk = (const float*)d_in[e0 + 2];
    const float* bk = (const float*)d_in[e0 + 3];
    const float* Wv = (const float*)d_in[e0 + 4];
    const float* bv = (const float*)d_in[e0 + 5];
    const float* Wo = (const float*)d_in[e0 + 6];
    const float* bo = (const float*)d_in[e0 + 7];
    const float* W1 = (const float*)d_in[e0 + 8];
    const float* b1 = (const float*)d_in[e0 + 9];
    const float* W2 = (const float*)d_in[e0 + 10];
    const float* b2 = (const float*)d_in[e0 + 11];
    const float* l1g = (const float*)d_in[e0 + 12];
    const float* l1b = (const float*)d_in[e0 + 13];
    const float* l2g = (const float*)d_in[e0 + 14];
    const float* l2b = (const float*)d_in[e0 + 15];

    for (int b = 0; b < Bc; ++b) {
      const float* xb = Z + (size_t)b * Sc * Dc;
      float* z2b = Z2 + (size_t)b * Sc * Dc;
      // projections
      launch_gemm(stream, xb, Wq, QB, bq, nullptr, Sc, Dc, Dc, Dc, Dc, Dc,
                  0, 0, 0, 0, 0, 0, 1, 1, 1, 1.0f, 0);
      launch_gemm(stream, xb, Wk, KB, bk, nullptr, Sc, Dc, Dc, Dc, Dc, Dc,
                  0, 0, 0, 0, 0, 0, 1, 1, 1, 1.0f, 0);
      launch_gemm(stream, xb, Wv, VB, bv, nullptr, Sc, Dc, Dc, Dc, Dc, Dc,
                  0, 0, 0, 0, 0, 0, 1, 1, 1, 1.0f, 0);
      // per-head scores = Q K^T / 8 over Dh=64 (head offset via sA2/sB2 = 64)
      launch_gemm(stream, QB, KB, ESC, nullptr, nullptr,
                  Sc, Sc, Dhc, Dc, Dc, Sc,
                  0, (long)Dhc, 0, (long)Dhc, 0, (long)Sc * Sc,
                  Henc, Henc, 1, 0.125f, 0);
      softmax_k<<<Henc * Sc, 256, 0, stream>>>(ESC, Sc, Sc, Henc * Sc);
      // ctx = A @ V, head-interleaved into [S, 512]
      launch_gemm(stream, ESC, VB, CTXB, nullptr, nullptr,
                  Sc, Dhc, Sc, Sc, Dc, Dc,
                  0, (long)Sc * Sc, 0, (long)Dhc, 0, (long)Dhc,
                  Henc, Henc, 0, 1.0f, 0);
      // attn out projection + residual -> Z2
      launch_gemm(stream, CTXB, Wo, z2b, bo, xb,
                  Sc, Dc, Dc, Dc, Dc, Dc,
                  0, 0, 0, 0, 0, 0, 1, 1, 1, 1.0f, 0);
    }
    layernorm_k<<<Bc * Sc, 256, 0, stream>>>(Z2, l1g, l1b, Z, Dc, Bc * Sc);
    for (int b = 0; b < Bc; ++b) {
      const float* xb = Z + (size_t)b * Sc * Dc;
      float* z2b = Z2 + (size_t)b * Sc * Dc;
      launch_gemm(stream, xb, W1, FFB, b1, nullptr,
                  Sc, FFc, Dc, Dc, Dc, FFc,
                  0, 0, 0, 0, 0, 0, 1, 1, 1, 1.0f, 1 /*gelu*/);
      launch_gemm(stream, FFB, W2, z2b, b2, xb,
                  Sc, Dc, FFc, FFc, FFc, Dc,
                  0, 0, 0, 0, 0, 0, 1, 1, 1, 1.0f, 0);
    }
    layernorm_k<<<Bc * Sc, 256, 0, stream>>>(Z2, l2g, l2b, Z, Dc, Bc * Sc);
  }

  // ---- final norm + projection + de-normalization ----
  layernorm_k<<<Bc * Sc, 256, 0, stream>>>(Z, norm_g, norm_b, Z2, Dc, Bc * Sc);
  launch_gemm(stream, Z2, proj_W, PO, proj_b, nullptr,
              Sc, PREDc, Dc, Dc, Dc, PREDc,
              (long)Sc * Dc, 0, 0, 0, (long)Sc * PREDc, 0,
              Bc, 1, 1, 1.0f, 0);
  {
    int tot = Bc * PREDc * Nc;
    final_k<<<cdiv(tot, 256), 256, 0, stream>>>(PO, MEAN, STDV, (float*)d_out, tot);
  }
}
